// KANMultiEncoderAttention_283467842571
// MI455X (gfx1250) — compile-verified, matrix operand order checked by reading, not run on hardware
//
#include <hip/hip_runtime.h>
#include <cmath>

// CDNA5 (gfx1250, wave32) implementation of KAN multi-encoder chunked attention.
// Flash-attention style (no materialized 4096x4096 attn), f16 WMMA w/ f32 acc,
// TDM (tensor_load_to_lds) double-buffered V-tile staging in the hot loop.

typedef __attribute__((ext_vector_type(16))) _Float16 v16h;
typedef __attribute__((ext_vector_type(8)))  float    v8f;
typedef __attribute__((ext_vector_type(4)))  unsigned int v4u;
typedef __attribute__((ext_vector_type(8)))  int      v8i;

#define IMW   128
#define HWSZ  16384   // 128*128
#define NPC   4096    // positions per 64x64 chunk
#define CDIM  256
#define DDIM  32
#define NE    3
#define NCK   4       // chunks per image (2x2)

// K-element index inside a 16-bit A/B fragment: vgpr j (0..7), lane half h (0..1).
// CDNA5 ISA "16-bit A-Matrix 16x32": v0..3 hold K in [0,16) split by half,
// v4..7 hold K in [16,32). Each vgpr packs K, K+1.
__device__ __forceinline__ int kmap(int j, int h) {
    return (j < 4) ? (8 * h + 2 * j) : (16 + 8 * h + 2 * (j - 4));
}

// chunk-local position n (0..4095) of chunk ck (0..3) -> linear image index h*128+w
__device__ __forceinline__ int chw(int ck, int n) {
    int kh = ck >> 1, kw = ck & 1;
    return ((kh << 6) + (n >> 6)) * IMW + (kw << 6) + (n & 63);
}

// ---------------------------------------------------------------------------
// TDM: async 2D tile load Global -> LDS. 2-byte elements. Uses only D# groups
// 0 and 1 (VADDR2/3 = NULL -> tensors up to 2D, CDNA5 ISA).
// tile_d0 contiguous elements per row, tile_d1 rows, row stride stride_d0.
// Tracked by TENSORcnt; caller pairs with s_wait_tensorcnt.
// ---------------------------------------------------------------------------
__device__ __forceinline__ void tdm_load_tile_2d(
    const void* gsrc, unsigned lds_off,
    unsigned tile_d0, unsigned tile_d1,
    unsigned tensor_d0, unsigned tensor_d1, unsigned stride_d0) {
    unsigned long long ga = (unsigned long long)(uintptr_t)gsrc;
    v4u g0;
    g0[0] = 1u;                                     // count=1 valid descriptor
    g0[1] = lds_off;                                // LDS byte address
    g0[2] = (unsigned)ga;                           // global_addr[31:0]
    g0[3] = ((unsigned)(ga >> 32) & 0x01FFFFFFu)    // global_addr[56:32]
            | (2u << 30);                           // type=2 ("image")
    v8i g1;
    g1[0] = (int)(1u << 16);                        // wg_mask=0, data_size=1 (2B)
    g1[1] = (int)((tensor_d0 & 0xFFFFu) << 16);     // tensor_dim0[15:0]
    g1[2] = (int)((tensor_d0 >> 16) | ((tensor_d1 & 0xFFFFu) << 16));
    g1[3] = (int)((tensor_d1 >> 16) | (tile_d0 << 16));   // tile_dim0
    g1[4] = (int)tile_d1;                           // tile_dim1; tile_dim2=0
    g1[5] = (int)stride_d0;                         // tensor_dim0_stride[31:0]
    g1[6] = 0;                                      // stride hi / dim1_stride
    g1[7] = 0;
    asm volatile("tensor_load_to_lds %0, %1" :: "s"(g0), "s"(g1) : "memory");
}

// ---------------------------------------------------------------------------
// Kernel 1: QKV projection. out[m,n] = sum_c W[m,c] * xc[c,n], f32->f16.
// One 16x16 output tile per wave, K looped 256 in steps of 32 (8 WMMAs).
// ---------------------------------------------------------------------------
__global__ __launch_bounds__(256) void qkv_kernel(
    const float* __restrict__ x0, const float* __restrict__ x1,
    const float* __restrict__ x2,
    const float* __restrict__ Wq, const float* __restrict__ Wk,
    const float* __restrict__ Wv,
    _Float16* __restrict__ qb, _Float16* __restrict__ kb,
    _Float16* __restrict__ vb) {
    const int lane = threadIdx.x & 31;
    const int wv   = threadIdx.x >> 5;
    const int i = lane & 15, h = lane >> 4;

    int tile = blockIdx.x * 8 + wv;            // 61440 tiles total
    const int per_ek = 512 + 512 + 4096;       // q,k,v tiles per (e,chunk)
    int ek  = tile / per_ek;
    int rem = tile - ek * per_ek;
    int e = ek >> 2, ck = ek & 3;

    const float* W; _Float16* ob; int mt, nt;
    if (rem < 512) {
        W = Wq + e * DDIM * CDIM; ob = qb + (size_t)ek * DDIM * NPC;
        mt = rem >> 8; nt = rem & 255;
    } else if (rem < 1024) {
        rem -= 512;
        W = Wk + e * DDIM * CDIM; ob = kb + (size_t)ek * DDIM * NPC;
        mt = rem >> 8; nt = rem & 255;
    } else {
        rem -= 1024;
        W = Wv + e * CDIM * CDIM; ob = vb + (size_t)ek * CDIM * NPC;
        mt = rem >> 8; nt = rem & 255;
    }
    const float* xp = (e == 0) ? x0 : (e == 1) ? x1 : x2;

    const int m0 = mt * 16, n0 = nt * 16;
    const int hwI = chw(ck, n0 + i);           // column n is fixed per lane

    v8f acc = {};
    for (int cb = 0; cb < CDIM; cb += 32) {
        if (cb + 32 < CDIM)                     // -> global_prefetch_b8
            __builtin_prefetch(xp + (size_t)(cb + 32) * HWSZ + hwI, 0, 1);
        v16h A, B;
#pragma unroll
        for (int j = 0; j < 8; ++j) {
            int kq = kmap(j, h);
            const float* wp = W + (size_t)(m0 + i) * CDIM + cb + kq;
            A[2 * j]     = (_Float16)wp[0];
            A[2 * j + 1] = (_Float16)wp[1];
            const float* bp = xp + (size_t)(cb + kq) * HWSZ + hwI;
            B[2 * j]     = (_Float16)bp[0];
            B[2 * j + 1] = (_Float16)bp[HWSZ];
        }
        acc = __builtin_amdgcn_wmma_f32_16x16x32_f16(false, A, false, B,
                                                     (short)0, acc, false, false);
    }
#pragma unroll
    for (int r = 0; r < 8; ++r)
        ob[(size_t)(m0 + r + 8 * h) * NPC + n0 + i] = (_Float16)acc[r];
}

// ---------------------------------------------------------------------------
// Kernel 2: flash attention per (e,chunk).  Block = 8 waves; wave (g, half):
// 16 query rows (nb = 64*blk + 16*g), 128 output channels (half*128..).
// m streamed in steps of 32. V tile staged by TDM (double-buffered, issued by
// wave 0 under a scalar branch), K tile staged cooperatively, P transposed
// through LDS between the score WMMA and the P*v^T WMMA.
// ---------------------------------------------------------------------------
__global__ __launch_bounds__(256) void attn_kernel(
    const float* __restrict__ x0, const float* __restrict__ x1,
    const float* __restrict__ x2,
    const _Float16* __restrict__ qb, const _Float16* __restrict__ kbuf,
    const _Float16* __restrict__ vb,
    const float* __restrict__ gamma, _Float16* __restrict__ ycat) {
    __shared__ _Float16 vt[2][CDIM * 32];   // v tile [c][m_local], 2 x 16 KB
    __shared__ _Float16 kt[32 * 32];        // k tile [m_local][d], 2 KB
    __shared__ _Float16 pt[8][16 * 32];     // per-wave P [n_local][m_local], 8 KB

    const int t    = threadIdx.x;
    const int lane = t & 31, wv = t >> 5;
    const int i = lane & 15, h = lane >> 4;
    const int ek = blockIdx.y;
    const int e = ek >> 2, ck = ek & 3;
    const int ngrp = wv >> 1, chalf = wv & 1;
    const int nb = blockIdx.x * 64 + ngrp * 16;

    const _Float16* q  = qb   + (size_t)ek * DDIM * NPC;
    const _Float16* kk = kbuf + (size_t)ek * DDIM * NPC;
    const _Float16* vv = vb   + (size_t)ek * CDIM * NPC;
    const float* xp = (e == 0) ? x0 : (e == 1) ? x1 : x2;

    // scalar (SGPR) wave-0 test so the TDM issue is a uniform branch:
    const bool w0 = __builtin_amdgcn_readfirstlane((int)threadIdx.x) < 32;

    // q fragment: A[n_local][d] = q[d][nb + n_local], loaded once (d == 32 == K)
    v16h aq;
#pragma unroll
    for (int j = 0; j < 8; ++j) {
        int kq = kmap(j, h);
        aq[2 * j]     = q[(size_t)kq * NPC + nb + i];
        aq[2 * j + 1] = q[(size_t)(kq + 1) * NPC + nb + i];
    }

    v8f oacc[8];
#pragma unroll
    for (int c = 0; c < 8; ++c) { v8f z = {}; oacc[c] = z; }
    float mrow[8], lrow[8];
#pragma unroll
    for (int r = 0; r < 8; ++r) { mrow[r] = -INFINITY; lrow[r] = 0.0f; }

    // TDM prologue: kick off V tile 0
    if (w0)
        tdm_load_tile_2d(vv, (unsigned)(uintptr_t)&vt[0][0],
                         32, CDIM, NPC, CDIM, NPC);

    for (int it = 0; it < NPC / 32; ++it) {
        const int m0 = it * 32, buf = it & 1;
        __syncthreads();   // everyone done with vt[buf^1] from iteration it-1
        if (w0) {
            if (it + 1 < NPC / 32) {
                // prefetch next V tile into the other buffer, then wait until
                // only it is outstanding => tile 'it' complete (TDM in-order)
                tdm_load_tile_2d(vv + (size_t)(m0 + 32),
                                 (unsigned)(uintptr_t)&vt[buf ^ 1][0],
                                 32, CDIM, NPC, CDIM, NPC);
                __builtin_amdgcn_s_wait_tensorcnt((short)1);
            } else {
                __builtin_amdgcn_s_wait_tensorcnt((short)0);
            }
        }
        { // stage k tile transposed: kt[m][d]
            int dd = t & 31, mq = (t >> 5) * 4;
            const _Float16* src = kk + (size_t)dd * NPC + m0 + mq;
#pragma unroll
            for (int j2 = 0; j2 < 4; ++j2) kt[(mq + j2) * 32 + dd] = src[j2];
        }
        __syncthreads();   // vt[buf] (TDM) and kt visible to all waves

        // scores: two 16x16 tiles, S[n][m] = sum_d q[d][n] k[d][m]
        v8f s[2];
#pragma unroll
        for (int tt = 0; tt < 2; ++tt) {
            v16h bk;
#pragma unroll
            for (int j = 0; j < 8; ++j) {
                int kq = kmap(j, h);
                bk[2 * j]     = kt[(tt * 16 + i) * 32 + kq];
                bk[2 * j + 1] = kt[(tt * 16 + i) * 32 + kq + 1];
            }
            v8f z = {};
            s[tt] = __builtin_amdgcn_wmma_f32_16x16x32_f16(false, aq, false, bk,
                                                           (short)0, z, false, false);
        }

        // online softmax: row r+8h of the D-frag lives across the 16 lanes of half h
#pragma unroll
        for (int r = 0; r < 8; ++r) {
            float lmax = fmaxf(s[0][r], s[1][r]);
            lmax = fmaxf(lmax, __shfl_xor(lmax, 1, 16));
            lmax = fmaxf(lmax, __shfl_xor(lmax, 2, 16));
            lmax = fmaxf(lmax, __shfl_xor(lmax, 4, 16));
            lmax = fmaxf(lmax, __shfl_xor(lmax, 8, 16));
            float mnew  = fmaxf(mrow[r], lmax);
            float scale = __expf(mrow[r] - mnew);
            float p0 = __expf(s[0][r] - mnew);
            float p1 = __expf(s[1][r] - mnew);
            float ps = p0 + p1;
            ps += __shfl_xor(ps, 1, 16);
            ps += __shfl_xor(ps, 2, 16);
            ps += __shfl_xor(ps, 4, 16);
            ps += __shfl_xor(ps, 8, 16);
            lrow[r] = lrow[r] * scale + ps;
            mrow[r] = mnew;
#pragma unroll
            for (int c = 0; c < 8; ++c) oacc[c][r] *= scale;
            // P -> LDS for layout change (D-layout -> A-layout)
            pt[wv][(r + 8 * h) * 32 + i]      = (_Float16)p0;
            pt[wv][(r + 8 * h) * 32 + 16 + i] = (_Float16)p1;
        }
        asm volatile("s_wait_dscnt 0x0" ::: "memory");  // intra-wave LDS RAW

        v16h ap;  // P as A fragment: A[n_local][m_local]
#pragma unroll
        for (int j = 0; j < 8; ++j) {
            int kq = kmap(j, h);
            ap[2 * j]     = pt[wv][i * 32 + kq];
            ap[2 * j + 1] = pt[wv][i * 32 + kq + 1];
        }
        // o[n][c] += sum_m P[n][m] v[c][m] : 8 channel tiles
#pragma unroll
        for (int c = 0; c < 8; ++c) {
            int c0 = chalf * 128 + c * 16;
            v16h bv;
#pragma unroll
            for (int j = 0; j < 8; ++j) {
                int kq = kmap(j, h);
                bv[2 * j]     = vt[buf][(c0 + i) * 32 + kq];
                bv[2 * j + 1] = vt[buf][(c0 + i) * 32 + kq + 1];
            }
            oacc[c] = __builtin_amdgcn_wmma_f32_16x16x32_f16(false, ap, false, bv,
                                                             (short)0, oacc[c], false, false);
        }
    }

    // epilogue: y = gamma*o/l + x, stored as cat[hw][e*256+c] (f16)
    const float g = gamma[e];
#pragma unroll
    for (int r = 0; r < 8; ++r) {
        float inv = 1.0f / lrow[r];
        int n = nb + r + 8 * h;
        int hwI = chw(ck, n);
#pragma unroll
        for (int c = 0; c < 8; ++c) {
            int cc = chalf * 128 + c * 16 + i;
            float xv  = xp[(size_t)cc * HWSZ + hwI];
            float val = g * oacc[c][r] * inv + xv;
            ycat[(size_t)hwI * (NE * CDIM) + e * CDIM + cc] = (_Float16)val;
        }
    }
}

// ---------------------------------------------------------------------------
// Kernel 3: out[o][hw] = sum_{c'=0..767} Wout[o][c'] * cat[hw][c'] (fp32 out)
// ---------------------------------------------------------------------------
__global__ __launch_bounds__(256) void out_kernel(
    const float* __restrict__ Wout, const _Float16* __restrict__ ycat,
    float* __restrict__ out) {
    const int lane = threadIdx.x & 31, wv = threadIdx.x >> 5;
    const int i = lane & 15, h = lane >> 4;
    int tile = blockIdx.x * 8 + wv;           // 16 x 1024 tiles
    int mt = tile >> 10, nt = tile & 1023;
    int m0 = mt * 16, n0 = nt * 16;

    v8f acc = {};
    for (int kbb = 0; kbb < NE * CDIM; kbb += 32) {
        if (kbb + 32 < NE * CDIM)
            __builtin_prefetch(ycat + (size_t)(n0 + i) * (NE * CDIM) + kbb + 32, 0, 1);
        v16h A, B;
#pragma unroll
        for (int j = 0; j < 8; ++j) {
            int kq = kmap(j, h);
            const float* wp = Wout + (size_t)(m0 + i) * (NE * CDIM) + kbb + kq;
            A[2 * j]     = (_Float16)wp[0];
            A[2 * j + 1] = (_Float16)wp[1];
            const _Float16* yp = ycat + (size_t)(n0 + i) * (NE * CDIM) + kbb + kq;
            B[2 * j]     = yp[0];
            B[2 * j + 1] = yp[1];
        }
        acc = __builtin_amdgcn_wmma_f32_16x16x32_f16(false, A, false, B,
                                                     (short)0, acc, false, false);
    }
#pragma unroll
    for (int r = 0; r < 8; ++r)
        out[(size_t)(m0 + r + 8 * h) * HWSZ + n0 + i] = acc[r];
}

// ---------------------------------------------------------------------------
extern "C" void kernel_launch(void* const* d_in, const int* in_sizes, int n_in,
                              void* d_out, int out_size, void* d_ws, size_t ws_size,
                              hipStream_t stream) {
    (void)in_sizes; (void)n_in; (void)out_size; (void)ws_size;
    const float* x0    = (const float*)d_in[0];
    const float* x1    = (const float*)d_in[1];
    const float* x2    = (const float*)d_in[2];
    const float* Wq    = (const float*)d_in[3];
    const float* Wk    = (const float*)d_in[4];
    const float* Wv    = (const float*)d_in[5];
    const float* gamma = (const float*)d_in[6];
    const float* Wout  = (const float*)d_in[7];

    // workspace: q (3 MB) | k (3 MB) | v (24 MB) | ycat (24 MB) -- all f16
    _Float16* qb   = (_Float16*)d_ws;
    _Float16* kb   = qb + (size_t)NE * NCK * DDIM * NPC;
    _Float16* vb   = kb + (size_t)NE * NCK * DDIM * NPC;
    _Float16* ycat = vb + (size_t)NE * NCK * CDIM * NPC;
    float* out = (float*)d_out;

    // 61440 tiles / 8 waves per block
    qkv_kernel<<<7680, 256, 0, stream>>>(x0, x1, x2, Wq, Wk, Wv, qb, kb, vb);
    // 12 (e,chunk) x 64 n-blocks of 64 rows
    attn_kernel<<<dim3(64, 12), 256, 0, stream>>>(x0, x1, x2, qb, kb, vb, gamma, ycat);
    // 16384 tiles / 8 waves per block
    out_kernel<<<2048, 256, 0, stream>>>(Wout, ycat, out);
}
